// TrajPromptGraphEncoder_40089224741070
// MI455X (gfx1250) — compile-verified, matrix operand order checked by reading, not run on hardware
//
#include <hip/hip_runtime.h>

typedef __bf16 bf16;
typedef __attribute__((ext_vector_type(16))) __bf16 v16bf;
typedef __attribute__((ext_vector_type(8)))  float  v8f;

#define DX    768
#define TT    2048
#define MW    32
#define NN    2080
#define DTRAJ 256

__device__ __forceinline__ float gelu_f(float x) {
  return 0.5f * x * (1.0f + erff(x * 0.70710678118654752440f));
}

// ---------------- elementwise converts ----------------
__global__ void k_f32_to_bf16(const float* __restrict__ in, bf16* __restrict__ out, int n) {
  int i = blockIdx.x * blockDim.x + threadIdx.x;
  if (i < n) out[i] = (bf16)in[i];
}

// W [K x Nc] row-major f32 -> Wt [Nc x K] bf16 (transposed, K-contiguous)
__global__ void k_transpose_w(const float* __restrict__ W, bf16* __restrict__ Wt, int K, int Nc) {
  int i = blockIdx.x * blockDim.x + threadIdx.x;
  if (i >= K * Nc) return;
  int k = i / Nc, n = i % Nc;
  Wt[(size_t)n * K + k] = (bf16)W[i];
}

__global__ void k_copy_words(const float* __restrict__ words, float* __restrict__ h) {
  int i = blockIdx.x * blockDim.x + threadIdx.x;
  if (i < MW * DX) h[(size_t)TT * DX + i] = words[i];
}

// ---------------- WMMA GEMM: C = A(bf16 MxK) * Wt(bf16 NxK)^T + bias ----------------
// block = 128 threads (4 waves). Block computes 128 rows x 64 cols; each wave a
// 32x64 strip (2 A fragments x 4 B fragments = 8 wmma per K-step).
// The 64xK bf16 weight panel is staged into LDS once per block with
// CDNA5 async global->LDS copies (ASYNCcnt), then read via ds_load_b128.
__global__ __launch_bounds__(128)
void k_gemm_bf16(const bf16* __restrict__ A, const bf16* __restrict__ Wt,
                 const float* __restrict__ bias,
                 float* __restrict__ outF, bf16* __restrict__ outB,
                 int Mr, int K, int doGelu) {
  extern __shared__ __align__(16) bf16 ldsB[];   // [64][K]
  const int Nc   = DX;
  const int wave = threadIdx.x >> 5;
  const int lane = threadIdx.x & 31;
  const int lm   = lane & 15;
  const int half = lane >> 4;
  const int rowBase = blockIdx.y * 128 + wave * 32;
  const int cb      = blockIdx.x * 64;

  // ---- async preload of the whole Wt panel (contiguous 64*K bf16) ----
  {
    const bf16* src = Wt + (size_t)cb * K;
    unsigned ldsBase = (unsigned)(uintptr_t)ldsB;      // low 32b of generic = LDS addr
    int nchunks = (64 * K) >> 3;                       // 16-byte chunks
    for (int c = threadIdx.x; c < nchunks; c += 128) {
      unsigned laddr = ldsBase + (unsigned)c * 16u;
      unsigned long long ga = (unsigned long long)(uintptr_t)src + (unsigned long long)c * 16ull;
      asm volatile("global_load_async_to_lds_b128 %0, %1, off"
                   :: "v"(laddr), "v"(ga) : "memory");
    }
    asm volatile("s_wait_asynccnt 0" ::: "memory");
    __syncthreads();
  }

  // A row pointers for the two 16-row fragments (clamped; stores guarded)
  int r0 = rowBase + lm;           if (r0 > Mr - 1) r0 = Mr - 1;
  int r1 = rowBase + 16 + lm;      if (r1 > Mr - 1) r1 = Mr - 1;
  const bf16* ap0 = A + (size_t)r0 * K;
  const bf16* ap1 = A + (size_t)r1 * K;

  v8f acc[2][4];
  for (int rt = 0; rt < 2; ++rt)
    for (int nt = 0; nt < 4; ++nt)
      for (int e = 0; e < 8; ++e) acc[rt][nt][e] = 0.0f;

  for (int k0 = 0; k0 < K; k0 += 32) {
    // A fragments: 16-bit A 16x32 layout -> two contiguous 16B chunks per lane
    union { v16bf v; uint4 q[2]; } a0, a1;
    a0.q[0] = *(const uint4*)(ap0 + k0 + half * 8);
    a0.q[1] = *(const uint4*)(ap0 + k0 + 16 + half * 8);
    a1.q[0] = *(const uint4*)(ap1 + k0 + half * 8);
    a1.q[1] = *(const uint4*)(ap1 + k0 + 16 + half * 8);
#pragma unroll
    for (int nt = 0; nt < 4; ++nt) {
      const bf16* bp = ldsB + (size_t)(nt * 16 + lm) * K + k0;
      union { v16bf v; uint4 q[2]; } bfr;
      bfr.q[0] = *(const uint4*)(bp + half * 8);
      bfr.q[1] = *(const uint4*)(bp + 16 + half * 8);
      acc[0][nt] = __builtin_amdgcn_wmma_f32_16x16x32_bf16(
          false, a0.v, false, bfr.v, (short)0, acc[0][nt], false, false);
      acc[1][nt] = __builtin_amdgcn_wmma_f32_16x16x32_bf16(
          false, a1.v, false, bfr.v, (short)0, acc[1][nt], false, false);
    }
  }

  // D layout: element (m = r + 8*half, n = lm) in VGPR r
#pragma unroll
  for (int rt = 0; rt < 2; ++rt) {
#pragma unroll
    for (int nt = 0; nt < 4; ++nt) {
      int n = cb + nt * 16 + lm;
      float bv = bias ? bias[n] : 0.0f;
#pragma unroll
      for (int r = 0; r < 8; ++r) {
        int row = rowBase + rt * 16 + r + half * 8;
        if (row < Mr) {
          float x = acc[rt][nt][r] + bv;
          if (doGelu) x = gelu_f(x);
          size_t idx = (size_t)row * Nc + n;
          if (outF) outF[idx] = x;
          if (outB) outB[idx] = (bf16)x;
        }
      }
    }
  }
}

// ---------------- LayerNorm (one row per block, 768 = 3*256) ----------------
__global__ __launch_bounds__(256)
void k_layernorm(const float* __restrict__ X, const float* __restrict__ g,
                 const float* __restrict__ b, const float* __restrict__ scale_ptr,
                 float* __restrict__ outF, bf16* __restrict__ outB, int rows) {
  int row = blockIdx.x;
  if (row >= rows) return;
  const float* xr = X + (size_t)row * DX;
  int tid = threadIdx.x;
  float v0 = xr[tid], v1 = xr[tid + 256], v2 = xr[tid + 512];
  __shared__ float red[256];
  red[tid] = v0 + v1 + v2;
  __syncthreads();
  for (int o = 128; o > 0; o >>= 1) { if (tid < o) red[tid] += red[tid + o]; __syncthreads(); }
  float mu = red[0] * (1.0f / DX);
  __syncthreads();
  float d0 = v0 - mu, d1 = v1 - mu, d2 = v2 - mu;
  red[tid] = d0 * d0 + d1 * d1 + d2 * d2;
  __syncthreads();
  for (int o = 128; o > 0; o >>= 1) { if (tid < o) red[tid] += red[tid + o]; __syncthreads(); }
  float rstd = rsqrtf(red[0] * (1.0f / DX) + 1e-5f);
  float sc = 1.0f;
  if (scale_ptr) sc = fminf(fmaxf(scale_ptr[0], 0.05f), 2.0f);

  float o0 = (d0 * rstd * g[tid]       + b[tid])       * sc;
  float o1 = (d1 * rstd * g[tid + 256] + b[tid + 256]) * sc;
  float o2 = (d2 * rstd * g[tid + 512] + b[tid + 512]) * sc;
  size_t base = (size_t)row * DX;
  if (outF) { outF[base + tid] = o0; outF[base + tid + 256] = o1; outF[base + tid + 512] = o2; }
  if (outB) { outB[base + tid] = (bf16)o0; outB[base + tid + 256] = (bf16)o1; outB[base + tid + 512] = (bf16)o2; }
}

// ---------------- word-vector norms ----------------
__global__ __launch_bounds__(32)
void k_wnorm(const float* __restrict__ words, float* __restrict__ wnorm) {
  int j = threadIdx.x;
  if (j >= MW) return;
  const float* wr = words + (size_t)j * DX;
  float s = 0.0f;
  for (int i = 0; i < DX; ++i) s += wr[i] * wr[i];
  wnorm[j] = fmaxf(sqrtf(s), 1e-12f);
}

// ---------------- alpha = softmax_j( cos(z0_t, w_j) ) , one wave per row ----------------
__global__ __launch_bounds__(32)
void k_alpha(const float* __restrict__ h, const float* __restrict__ words,
             const float* __restrict__ wnorm, const int* __restrict__ valid_ptr,
             float* __restrict__ alpha) {
  int t = blockIdx.x;
  int valid = valid_ptr[0];
  if (t >= valid) return;
  int j = threadIdx.x;
  const float* zr = h + (size_t)t * DX;
  float s = 0.0f;
  for (int i = j; i < DX; i += 32) { float x = zr[i]; s += x * x; }
  for (int o = 16; o > 0; o >>= 1) s += __shfl_xor(s, o, 32);
  float zn = fmaxf(sqrtf(s), 1e-12f);
  const float* wr = words + (size_t)j * DX;
  float d = 0.0f;
  for (int i = 0; i < DX; ++i) d += zr[i] * wr[i];
  float logit = d / (zn * wnorm[j]);          // TAU == 1
  float mx = logit;
  for (int o = 16; o > 0; o >>= 1) mx = fmaxf(mx, __shfl_xor(mx, o, 32));
  float e = __expf(logit - mx);
  float se = e;
  for (int o = 16; o > 0; o >>= 1) se += __shfl_xor(se, o, 32);
  alpha[(size_t)t * MW + j] = e / se;
}

// ---------------- beta / inv_beta (deterministic sequential reduction) ----------------
__global__ __launch_bounds__(32)
void k_beta(const float* __restrict__ alpha, const int* __restrict__ valid_ptr,
            float* __restrict__ beta, float* __restrict__ invb) {
  int j = threadIdx.x;
  if (j >= MW) return;
  int valid = valid_ptr[0];
  float s = 0.0f;
  for (int t = 0; t < valid; ++t) s += alpha[(size_t)t * MW + j];
  beta[j] = s;
  float iv = 1.0f / (s + 1e-6f);
  invb[j] = fminf(fmaxf(iv, 0.2f), 5.0f);
}

// ---------------- GCN residual update: traj nodes (temporal chain) ----------------
__global__ void k_update_traj(const float* __restrict__ m, const int* __restrict__ valid_ptr,
                              float* __restrict__ h) {
  int i = blockIdx.x * blockDim.x + threadIdx.x;
  if (i >= TT * DX) return;
  int row = i / DX;
  int valid = valid_ptr[0];
  float out = 0.0f;
  if (row >= 1 && row < valid) out = m[i - DX];
  h[i] += gelu_f(out);
}

// ---------------- GCN residual update: word nodes (alpha^T @ m) ----------------
__global__ __launch_bounds__(256)
void k_update_word(const float* __restrict__ m, const float* __restrict__ alpha,
                   const float* __restrict__ beta, const float* __restrict__ invb,
                   const int* __restrict__ valid_ptr, float* __restrict__ h) {
  int j = blockIdx.x;
  int c = threadIdx.x;
  int valid = valid_ptr[0];
  float a0 = 0.0f, a1 = 0.0f, a2 = 0.0f;
  for (int t = 0; t < valid; ++t) {
    float a = alpha[(size_t)t * MW + j];
    const float* mr = m + (size_t)t * DX;
    a0 += a * mr[c];
    a1 += a * mr[c + 256];
    a2 += a * mr[c + 512];
  }
  float s = invb[j] / fmaxf(beta[j], 1.0f);
  size_t base = (size_t)(TT + j) * DX;
  h[base + c]       += gelu_f(a0 * s);
  h[base + c + 256] += gelu_f(a1 * s);
  h[base + c + 512] += gelu_f(a2 * s);
}

extern "C" void kernel_launch(void* const* d_in, const int* in_sizes, int n_in,
                              void* d_out, int out_size, void* d_ws, size_t ws_size,
                              hipStream_t stream) {
  (void)in_sizes; (void)n_in; (void)out_size; (void)ws_size;
  const float* traj       = (const float*)d_in[0];
  const float* words      = (const float*)d_in[1];
  const int*   valid      = (const int*)d_in[4];   // low dword of the scalar
  const float* W1         = (const float*)d_in[5];
  const float* b1         = (const float*)d_in[6];
  const float* W2         = (const float*)d_in[7];
  const float* b2         = (const float*)d_in[8];
  const float* ln_g       = (const float*)d_in[9];
  const float* ln_b       = (const float*)d_in[10];
  const float* traj_scale = (const float*)d_in[11];
  const float* g_ln_g[2]  = { (const float*)d_in[12], (const float*)d_in[16] };
  const float* g_ln_b[2]  = { (const float*)d_in[13], (const float*)d_in[17] };
  const float* g_W[2]     = { (const float*)d_in[14], (const float*)d_in[18] };
  const float* g_b[2]     = { (const float*)d_in[15], (const float*)d_in[19] };
  float* h = (float*)d_out;

  char* ws = (char*)d_ws;
  size_t off = 0;
  auto alloc = [&](size_t bytes) -> void* {
    void* p = ws + off;
    off = (off + bytes + 255) & ~(size_t)255;
    return p;
  };
  bf16*  W1t   = (bf16*)alloc((size_t)DX * DTRAJ * sizeof(bf16));
  bf16*  W2t   = (bf16*)alloc((size_t)DX * DX * sizeof(bf16));
  bf16*  g0Wt  = (bf16*)alloc((size_t)DX * DX * sizeof(bf16));
  bf16*  g1Wt  = (bf16*)alloc((size_t)DX * DX * sizeof(bf16));
  bf16*  Abuf  = (bf16*)alloc((size_t)NN * DX * sizeof(bf16));
  bf16*  t1b   = (bf16*)alloc((size_t)TT * DX * sizeof(bf16));
  float* z     = (float*)alloc((size_t)TT * DX * sizeof(float));
  float* mbuf  = (float*)alloc((size_t)NN * DX * sizeof(float));
  float* alpha = (float*)alloc((size_t)TT * MW * sizeof(float));
  float* wnorm = (float*)alloc(256);
  float* beta  = (float*)alloc(256);
  float* invb  = (float*)alloc(256);
  bf16* gWt[2] = { g0Wt, g1Wt };

  // weight conversions (bf16, transposed for K-contiguous B fragments)
  k_transpose_w<<<(DTRAJ * DX + 255) / 256, 256, 0, stream>>>(W1, W1t, DTRAJ, DX);
  k_transpose_w<<<(DX * DX + 255) / 256, 256, 0, stream>>>(W2, W2t, DX, DX);
  k_transpose_w<<<(DX * DX + 255) / 256, 256, 0, stream>>>(g_W[0], g0Wt, DX, DX);
  k_transpose_w<<<(DX * DX + 255) / 256, 256, 0, stream>>>(g_W[1], g1Wt, DX, DX);
  // traj tokens -> bf16 (Abuf viewed as [TT x DTRAJ])
  k_f32_to_bf16<<<(TT * DTRAJ + 255) / 256, 256, 0, stream>>>(traj, Abuf, TT * DTRAJ);

  // t1 = gelu(traj @ W1 + b1) -> bf16
  {
    dim3 g(DX / 64, (TT + 127) / 128);
    size_t shm = (size_t)64 * DTRAJ * sizeof(bf16);
    k_gemm_bf16<<<g, 128, shm, stream>>>(Abuf, W1t, b1, nullptr, t1b, TT, DTRAJ, 1);
  }
  // z = t1 @ W2 + b2 -> f32
  {
    dim3 g(DX / 64, (TT + 127) / 128);
    size_t shm = (size_t)64 * DX * sizeof(bf16);
    k_gemm_bf16<<<g, 128, shm, stream>>>(t1b, W2t, b2, z, nullptr, TT, DX, 0);
  }
  // z0 = LN(z) * clip(traj_scale) -> h[0:TT)
  k_layernorm<<<TT, 256, 0, stream>>>(z, ln_g, ln_b, traj_scale, h, nullptr, TT);
  // h[TT:NN) = word tokens
  k_copy_words<<<(MW * DX + 255) / 256, 256, 0, stream>>>(words, h);
  // attention weights
  k_wnorm<<<1, 32, 0, stream>>>(words, wnorm);
  k_alpha<<<TT, 32, 0, stream>>>(h, words, wnorm, valid, alpha);
  k_beta<<<1, 32, 0, stream>>>(alpha, valid, beta, invb);

  for (int l = 0; l < 2; ++l) {
    k_layernorm<<<NN, 256, 0, stream>>>(h, g_ln_g[l], g_ln_b[l], nullptr, nullptr, Abuf, NN);
    dim3 g(DX / 64, (NN + 127) / 128);
    size_t shm = (size_t)64 * DX * sizeof(bf16);
    k_gemm_bf16<<<g, 128, shm, stream>>>(Abuf, gWt[l], g_b[l], mbuf, nullptr, NN, DX, 0);
    k_update_traj<<<(TT * DX + 255) / 256, 256, 0, stream>>>(mbuf, valid, h);
    k_update_word<<<MW, 256, 0, stream>>>(mbuf, alpha, beta, invb, valid, h);
  }
}
